// PointTransformerLayer_46351287059060
// MI455X (gfx1250) — matrix operand order repaired and checked by reading, not compiled, not run on hardware
//
#include <hip/hip_runtime.h>

typedef __attribute__((ext_vector_type(16))) __bf16 v16bf;
typedef __attribute__((ext_vector_type(8)))  float  v8f;
typedef __attribute__((ext_vector_type(4)))  unsigned tdm_g0_t;
typedef __attribute__((ext_vector_type(8)))  int      tdm_g1_t;
typedef __attribute__((ext_vector_type(4)))  int      tdm_g2_t;

__device__ __forceinline__ __bf16 f2bf(float f) { return (__bf16)f; }

// ---------------------------------------------------------------------------
// Tensor Data Mover: DMA a 2D f32 tile (width x height elems, row stride in
// elems) from global memory into LDS at lds_off. Issued per-wave (EXEC is
// ignored by TDM); completion via TENSORcnt.
// ---------------------------------------------------------------------------
__device__ __forceinline__ void tdm_load_2d_f32(const void* gptr, unsigned lds_off,
                                                unsigned width, unsigned height,
                                                unsigned stride) {
  unsigned long long ga = (unsigned long long)(uintptr_t)gptr;
  tdm_g0_t g0;
  g0[0] = 1u;                                             // count=1, user D#
  g0[1] = lds_off;                                        // LDS byte address
  g0[2] = (unsigned)(ga & 0xffffffffu);                   // global_addr[31:0]
  g0[3] = (unsigned)((ga >> 32) & 0x01ffffffu)            // global_addr[56:32]
        | 0x80000000u;                                    // type=2 (image)
  tdm_g1_t g1;
  g1[0] = 0x00020000;                 // workgroup_mask=0, data_size=4B (code 2)
  g1[1] = (int)(width << 16);         // tensor_dim0[15:0]  @ bits 63:48
  g1[2] = (int)(height << 16);        // tensor_dim1[15:0]  @ bits 95:80
  g1[3] = (int)(width << 16);         // tile_dim0          @ bits 127:112
  g1[4] = (int)height;                // tile_dim1 (tile_dim2 = 0)
  g1[5] = (int)stride;                // tensor_dim0_stride[31:0]
  g1[6] = 0;                          // dim0_stride hi, dim1_stride lo
  g1[7] = 0;
  tdm_g2_t gz4 = {0, 0, 0, 0};
  tdm_g1_t gz8 = {0, 0, 0, 0, 0, 0, 0, 0};
  __builtin_amdgcn_tensor_load_to_lds(g0, g1, gz4, gz4, gz8, 0);
}

// ---------------------------------------------------------------------------
// GEMM 1: x_q = x^T @ wq + b    A(n,c)=x[b][c][n] (column-strided), 2048x64x32
// one wave = one 16x16 D tile, two K=32 WMMA steps; W staged to LDS via TDM.
// ---------------------------------------------------------------------------
__global__ void gemm_xq_kernel(const float* __restrict__ x,
                               const float* __restrict__ w,
                               const float* __restrict__ bias,
                               float* __restrict__ out) {
  __shared__ float wtile[64 * 32];
  if (threadIdx.x < 32) {
    tdm_load_2d_f32(w, (unsigned)(uintptr_t)wtile, 32u, 64u, 32u);
    __builtin_amdgcn_s_wait_tensorcnt(0);
  }
  __syncthreads();
  int wave = (int)((blockIdx.x * blockDim.x + threadIdx.x) >> 5);
  int lane = (int)(threadIdx.x & 31);
  int nt = wave & 1;            // 2 N tiles
  int mt = (wave >> 1) & 127;   // 128 M tiles
  int b  = wave >> 8;           // 4 batches
  const float* A = x + (size_t)b * 64 * 2048;
  int hh    = lane >> 4;        // half-wave
  int m     = lane & 15;
  int khalf = hh << 3;
  int koff  = hh << 4;
  int row   = mt * 16 + m;
  v8f acc = {};
#pragma unroll
  for (int k0 = 0; k0 < 64; k0 += 32) {
    v16bf af, bfr;
#pragma unroll
    for (int e = 0; e < 8; ++e) {
      af[e]     = f2bf(A[(k0 + khalf + e) * 2048 + row]);
      af[e + 8] = f2bf(A[(k0 + 16 + khalf + e) * 2048 + row]);
    }
#pragma unroll
    for (int e = 0; e < 16; ++e)
      bfr[e] = f2bf(wtile[(k0 + khalf + (e & 7) + ((e >> 3) << 4)) * 32 + nt * 16 + m]);
    acc = __builtin_amdgcn_wmma_f32_16x16x32_bf16(false, af, false, bfr,
                                                  (short)0, acc, false, false);
  }
  float bv = bias[nt * 16 + m];
  float* O = out + (size_t)b * 2048 * 32;
#pragma unroll
  for (int i = 0; i < 8; ++i) {
    int M = mt * 16 + (hh << 3) + i;
    O[M * 32 + nt * 16 + m] = acc[i] + bv;
  }
}

// ---------------------------------------------------------------------------
// GEMM 2/3: x1 (B,8192,64) row-major @ W(64,NC) + b,  NC = 32 (wk) or 64 (wv)
// ---------------------------------------------------------------------------
__global__ void gemm_x1_kernel(const float* __restrict__ x1,
                               const float* __restrict__ w,
                               const float* __restrict__ bias,
                               float* __restrict__ out, int ntshift) {
  __shared__ float wtile[64 * 64];
  int NC = 16 << ntshift;
  if (threadIdx.x < 32) {
    tdm_load_2d_f32(w, (unsigned)(uintptr_t)wtile, (unsigned)NC, 64u, (unsigned)NC);
    __builtin_amdgcn_s_wait_tensorcnt(0);
  }
  __syncthreads();
  int wave = (int)((blockIdx.x * blockDim.x + threadIdx.x) >> 5);
  int lane = (int)(threadIdx.x & 31);
  int nt   = wave & ((1 << ntshift) - 1);
  int rest = wave >> ntshift;
  int mt   = rest & 511;        // 512 M tiles
  int b    = rest >> 9;
  int hh    = lane >> 4;
  int m     = lane & 15;
  int khalf = hh << 3;
  int koff  = hh << 4;
  const float* Arow = x1 + ((size_t)b * 8192 + (size_t)(mt * 16 + m)) * 64;
  v8f acc = {};
#pragma unroll
  for (int k0 = 0; k0 < 64; k0 += 32) {
    v16bf af, bfr;
#pragma unroll
    for (int e = 0; e < 8; ++e) {
      af[e]     = f2bf(Arow[k0 + khalf + e]);
      af[e + 8] = f2bf(Arow[k0 + 16 + khalf + e]);
    }
#pragma unroll
    for (int e = 0; e < 16; ++e)
      bfr[e] = f2bf(wtile[(k0 + khalf + (e & 7) + ((e >> 3) << 4)) * NC + nt * 16 + m]);
    acc = __builtin_amdgcn_wmma_f32_16x16x32_bf16(false, af, false, bfr,
                                                  (short)0, acc, false, false);
  }
  float bv = bias[nt * 16 + m];
  float* O = out + (size_t)b * 8192 * NC;
#pragma unroll
  for (int i = 0; i < 8; ++i) {
    int M = mt * 16 + (hh << 3) + i;
    O[(size_t)M * NC + nt * 16 + m] = acc[i] + bv;
  }
}

// ---------------------------------------------------------------------------
// Farthest point sampling: one block per batch; 16 pts/thread in registers,
// running min-distance in registers, argmax via LDS tree reduction.
// ---------------------------------------------------------------------------
__global__ __launch_bounds__(512) void fps_kernel(const float* __restrict__ xyz1,
                                                  float* __restrict__ new_xyz) {
  __shared__ float sdist[512];
  __shared__ int   sidx[512];
  __shared__ float bc[3];
  int b = blockIdx.x;
  int t = threadIdx.x;
  const float* P = xyz1 + (size_t)b * 8192 * 3;
  float px[16], py[16], pz[16], dd[16];
#pragma unroll
  for (int j = 0; j < 16; ++j) {
    int p = j * 512 + t;
    px[j] = P[p * 3 + 0];
    py[j] = P[p * 3 + 1];
    pz[j] = P[p * 3 + 2];
    dd[j] = 1e10f;
  }
  if (t == 0) { bc[0] = P[0]; bc[1] = P[1]; bc[2] = P[2]; }
  for (int s = 0; s < 2048; ++s) {
    __syncthreads();
    float cx = bc[0], cy = bc[1], cz = bc[2];
    if (t == 0) {
      float* o = new_xyz + ((size_t)b * 2048 + s) * 3;
      o[0] = cx; o[1] = cy; o[2] = cz;
    }
    float best = -1.0f; int bi = 0x7fffffff;
#pragma unroll
    for (int j = 0; j < 16; ++j) {
      float ex = px[j] - cx, ey = py[j] - cy, ez = pz[j] - cz;
      float d2 = ex * ex + ey * ey + ez * ez;
      dd[j] = fminf(dd[j], d2);
      int p = j * 512 + t;
      if (dd[j] > best || (dd[j] == best && p < bi)) { best = dd[j]; bi = p; }
    }
    sdist[t] = best; sidx[t] = bi;
    __syncthreads();
    for (int off = 256; off > 0; off >>= 1) {
      if (t < off) {
        float od = sdist[t + off]; int oi = sidx[t + off];
        if (od > sdist[t] || (od == sdist[t] && oi < sidx[t])) {
          sdist[t] = od; sidx[t] = oi;
        }
      }
      __syncthreads();
    }
    if (t == 0) {
      int far = sidx[0];                 // re-read winner coords from global
      bc[0] = P[far * 3 + 0];
      bc[1] = P[far * 3 + 1];
      bc[2] = P[far * 3 + 2];
    }
  }
}

// ---------------------------------------------------------------------------
// Ball query: one wave per group; ballot scan yields first 16 ascending hits.
// ---------------------------------------------------------------------------
__global__ void ball_kernel(const float* __restrict__ xyz1,
                            const float* __restrict__ new_xyz,
                            int* __restrict__ idx) {
  int wave = (int)((blockIdx.x * blockDim.x + threadIdx.x) >> 5);
  int lane = (int)(threadIdx.x & 31);
  int grp = wave;                 // b*2048 + s
  int b   = wave >> 11;
  const float* P = xyz1 + (size_t)b * 8192 * 3;
  const float* C = new_xyz + (size_t)grp * 3;
  float cx = C[0], cy = C[1], cz = C[2];
  int* out = idx + grp * 16;
  int count = 0;
  int firstIdx = 0;
  bool haveFirst = false;
  for (int base = 0; base < 8192 && count < 16; base += 32) {
    int p = base + lane;
    float ex = P[p * 3 + 0] - cx;
    float ey = P[p * 3 + 1] - cy;
    float ez = P[p * 3 + 2] - cz;
    bool hit = (ex * ex + ey * ey + ez * ez) <= 0.04f;   // r^2
    unsigned mask  = __builtin_amdgcn_ballot_w32(hit);
    unsigned below = mask & ((1u << lane) - 1u);
    int pos = count + (int)__builtin_popcount(below);
    if (hit && pos < 16) out[pos] = p;
    if (!haveFirst && mask != 0u) {
      firstIdx = base + (int)__builtin_ctz(mask);
      haveFirst = true;
    }
    count += (int)__builtin_popcount(mask);
  }
  int cnt = count < 16 ? count : 16;
  if (lane < 16 && lane >= cnt) out[lane] = firstIdx;
}

// ---------------------------------------------------------------------------
// shared 3->3 dense + LN(3) + relu helper (p_l1/p_ln and p1_l1/p1_ln blocks)
// ---------------------------------------------------------------------------
__device__ __forceinline__ void pmlp3(float i0, float i1, float i2,
                                      const float* __restrict__ w,
                                      const float* __restrict__ bias,
                                      const float* __restrict__ g,
                                      const float* __restrict__ bt,
                                      float h[3]) {
  float a0 = i0 * w[0] + i1 * w[3] + i2 * w[6] + bias[0];
  float a1 = i0 * w[1] + i1 * w[4] + i2 * w[7] + bias[1];
  float a2 = i0 * w[2] + i1 * w[5] + i2 * w[8] + bias[2];
  float mu = (a0 + a1 + a2) * (1.0f / 3.0f);
  float d0 = a0 - mu, d1 = a1 - mu, d2 = a2 - mu;
  float inv = rsqrtf((d0 * d0 + d1 * d1 + d2 * d2) * (1.0f / 3.0f) + 1e-5f);
  h[0] = fmaxf(d0 * inv * g[0] + bt[0], 0.0f);
  h[1] = fmaxf(d1 * inv * g[1] + bt[1], 0.0f);
  h[2] = fmaxf(d2 * inv * g[2] + bt[2], 0.0f);
}

// p_r_1 = dense(relu(LN(dense(xyz^T, p1_l1))), p1_l2)  -> (B,2048,32)
__global__ void pr1_kernel(const float* __restrict__ xyz,
                           const float* __restrict__ w1, const float* __restrict__ b1,
                           const float* __restrict__ g,  const float* __restrict__ bt,
                           const float* __restrict__ w2, const float* __restrict__ b2,
                           float* __restrict__ pr1) {
  int i = blockIdx.x * blockDim.x + threadIdx.x;
  if (i >= 4 * 2048) return;
  int b = i >> 11, n = i & 2047;
  const float* X = xyz + (size_t)b * 3 * 2048;
  float h[3];
  pmlp3(X[n], X[2048 + n], X[4096 + n], w1, b1, g, bt, h);
#pragma unroll
  for (int m = 0; m < 32; ++m)
    pr1[(size_t)i * 32 + m] =
        h[0] * w2[m] + h[1] * w2[32 + m] + h[2] * w2[64 + m] + b2[m];
}

// ---------------------------------------------------------------------------
// Per (group, sample) fused scoring MLP -> attention logits (B,2048,16,8)
// ---------------------------------------------------------------------------
__global__ void logits_kernel(const float* __restrict__ xq,
                              const float* __restrict__ pr1,
                              const float* __restrict__ xkf,
                              const float* __restrict__ xyz1,
                              const float* __restrict__ newxyz,
                              const int* __restrict__ ballidx,
                              const float* __restrict__ pl1w, const float* __restrict__ pl1b,
                              const float* __restrict__ plng, const float* __restrict__ plnb,
                              const float* __restrict__ pl2w, const float* __restrict__ pl2b,
                              const float* __restrict__ ln1g, const float* __restrict__ ln1b,
                              const float* __restrict__ w1w,  const float* __restrict__ w1b,
                              const float* __restrict__ ln2g, const float* __restrict__ ln2b,
                              const float* __restrict__ w2w,  const float* __restrict__ w2b,
                              float* __restrict__ logits) {
  int gid = blockIdx.x * blockDim.x + threadIdx.x;   // (b*2048+s)*16 + t
  if (gid >= 4 * 2048 * 16) return;
  int t   = gid & 15;
  int grp = gid >> 4;
  int b   = grp >> 11;
  int gi  = ballidx[grp * 16 + t];
  const float* Pt = xyz1 + ((size_t)b * 8192 + gi) * 3;
  const float* C  = newxyz + (size_t)grp * 3;
  float h[3];
  pmlp3(Pt[0] - C[0], Pt[1] - C[1], Pt[2] - C[2], pl1w, pl1b, plng, plnb, h);
  const float* xk = xkf + ((size_t)b * 8192 + gi) * 32;
  const float* q  = xq  + (size_t)grp * 32;
  const float* p1 = pr1 + (size_t)grp * 32;
  float r[32];
  float sum = 0.0f, sq = 0.0f;
#pragma unroll
  for (int m = 0; m < 32; ++m) {
    // p_r_sum[m] = p_r[m] + p_r[32+m], p_l2 is (3,64) row-major
    float pa = h[0] * pl2w[m]      + h[1] * pl2w[64 + m] + h[2] * pl2w[128 + m] + pl2b[m];
    float pb = h[0] * pl2w[32 + m] + h[1] * pl2w[96 + m] + h[2] * pl2w[160 + m] + pl2b[32 + m];
    float rv = xk[m] - (q[m] + p1[m]) + (pa + pb);
    r[m] = rv; sum += rv; sq += rv * rv;
  }
  float mu  = sum * (1.0f / 32.0f);
  float var = sq * (1.0f / 32.0f) - mu * mu;
  float inv = rsqrtf(var + 1e-5f);
  float h2[8];
#pragma unroll
  for (int j = 0; j < 8; ++j) h2[j] = w1b[j];
#pragma unroll
  for (int m = 0; m < 32; ++m) {
    float hn = fmaxf((r[m] - mu) * inv * ln1g[m] + ln1b[m], 0.0f);
#pragma unroll
    for (int j = 0; j < 8; ++j) h2[j] += hn * w1w[m * 8 + j];
  }
  float s2 = 0.0f, q2 = 0.0f;
#pragma unroll
  for (int j = 0; j < 8; ++j) { s2 += h2[j]; q2 += h2[j] * h2[j]; }
  float mu2  = s2 * 0.125f;
  float inv2 = rsqrtf(q2 * 0.125f - mu2 * mu2 + 1e-5f);
  float h3[8];
#pragma unroll
  for (int j = 0; j < 8; ++j)
    h3[j] = fmaxf((h2[j] - mu2) * inv2 * ln2g[j] + ln2b[j], 0.0f);
  float* L = logits + (size_t)gid * 8;
#pragma unroll
  for (int i = 0; i < 8; ++i) {
    float v = w2b[i];
#pragma unroll
    for (int j = 0; j < 8; ++j) v += h3[j] * w2w[j * 8 + i];
    L[i] = v;
  }
}

// ---------------------------------------------------------------------------
// Softmax over axis=1 (the 2048 DOWN dim): 512 columns (b,t,i), max+sumexp.
// ---------------------------------------------------------------------------
__global__ void softmax_stats_kernel(const float* __restrict__ logits,
                                     float* __restrict__ smax,
                                     float* __restrict__ ssum) {
  __shared__ float red[256];
  int col = blockIdx.x;          // b*128 + t*8 + i
  int b   = col >> 7;
  int ti  = col & 127;
  const float* L = logits + (size_t)b * 2048 * 128 + ti;
  int t = threadIdx.x;
  float m = -3.4e38f;
  for (int s = t; s < 2048; s += 256) m = fmaxf(m, L[(size_t)s * 128]);
  red[t] = m; __syncthreads();
  for (int off = 128; off > 0; off >>= 1) {
    if (t < off) red[t] = fmaxf(red[t], red[t + off]);
    __syncthreads();
  }
  float gmax = red[0];
  __syncthreads();
  float sum = 0.0f;
  for (int s = t; s < 2048; s += 256) sum += __expf(L[(size_t)s * 128] - gmax);
  red[t] = sum; __syncthreads();
  for (int off = 128; off > 0; off >>= 1) {
    if (t < off) red[t] += red[t + off];
    __syncthreads();
  }
  if (t == 0) { smax[col] = gmax; ssum[col] = red[0]; }
}

// ---------------------------------------------------------------------------
// Final: recompute p_r (3->64) per sample, gather x_v, weighted sum -> out.
// 64 threads = 64 output channels per group; einsum 'bntsi,bnti->bnsi'.
// ---------------------------------------------------------------------------
__global__ __launch_bounds__(64)
void output_kernel(const float* __restrict__ xvf,
                   const int* __restrict__ ballidx,
                   const float* __restrict__ xyz1,
                   const float* __restrict__ newxyz,
                   const float* __restrict__ pl1w, const float* __restrict__ pl1b,
                   const float* __restrict__ plng, const float* __restrict__ plnb,
                   const float* __restrict__ pl2w, const float* __restrict__ pl2b,
                   const float* __restrict__ logits,
                   const float* __restrict__ smax,
                   const float* __restrict__ ssum,
                   float* __restrict__ out) {
  __shared__ float sg[16][3];
  __shared__ float sw[16][8];
  __shared__ int   sgi[16];
  int grp = blockIdx.x;          // b*2048 + s
  int b   = grp >> 11;
  int c   = (int)threadIdx.x;    // 0..63
  if (c < 16) {
    int t = c;
    int gi = ballidx[grp * 16 + t];
    sgi[t] = gi;
    const float* Pt = xyz1 + ((size_t)b * 8192 + gi) * 3;
    const float* C  = newxyz + (size_t)grp * 3;
    float h[3];
    pmlp3(Pt[0] - C[0], Pt[1] - C[1], Pt[2] - C[2], pl1w, pl1b, plng, plnb, h);
    sg[t][0] = h[0]; sg[t][1] = h[1]; sg[t][2] = h[2];
    const float* L = logits + ((size_t)grp * 16 + t) * 8;
#pragma unroll
    for (int i = 0; i < 8; ++i) {
      int col = b * 128 + t * 8 + i;
      sw[t][i] = __expf(L[i] - smax[col]) / ssum[col];
    }
  }
  __syncthreads();
  float w0 = pl2w[c], w1 = pl2w[64 + c], w2 = pl2w[128 + c], bb = pl2b[c];
  int i8 = c & 7;
  float acc = 0.0f;
#pragma unroll 4
  for (int t = 0; t < 16; ++t) {
    float pr = sg[t][0] * w0 + sg[t][1] * w1 + sg[t][2] * w2 + bb;
    float v  = xvf[((size_t)b * 8192 + sgi[t]) * 64 + c] + pr;
    acc += v * sw[t][i8];
  }
  out[(size_t)grp * 64 + c] = acc;
}

// ---------------------------------------------------------------------------
// Launcher: inputs in setup_inputs() dict (insertion) order, params leaves
// flattened depth-first (w, b) per linear, (gamma, beta) per LN.
// ---------------------------------------------------------------------------
extern "C" void kernel_launch(void* const* d_in, const int* in_sizes, int n_in,
                              void* d_out, int out_size, void* d_ws, size_t ws_size,
                              hipStream_t stream) {
  (void)in_sizes; (void)n_in; (void)out_size; (void)ws_size;
  const float* x     = (const float*)d_in[0];   // (4,64,2048)
  const float* xyz   = (const float*)d_in[1];   // (4,3,2048)
  const float* x1    = (const float*)d_in[2];   // (4,8192,64)
  const float* xyz1  = (const float*)d_in[3];   // (4,8192,3)
  const float* wq_w  = (const float*)d_in[4];
  const float* wq_b  = (const float*)d_in[5];
  const float* wk_w  = (const float*)d_in[6];
  const float* wk_b  = (const float*)d_in[7];
  const float* wv_w  = (const float*)d_in[8];
  const float* wv_b  = (const float*)d_in[9];
  const float* pl1w  = (const float*)d_in[10];
  const float* pl1b  = (const float*)d_in[11];
  const float* plng  = (const float*)d_in[12];
  const float* plnb  = (const float*)d_in[13];
  const float* pl2w  = (const float*)d_in[14];
  const float* pl2b  = (const float*)d_in[15];
  const float* p1l1w = (const float*)d_in[16];
  const float* p1l1b = (const float*)d_in[17];
  const float* p1lng = (const float*)d_in[18];
  const float* p1lnb = (const float*)d_in[19];
  const float* p1l2w = (const float*)d_in[20];
  const float* p1l2b = (const float*)d_in[21];
  const float* ln1g  = (const float*)d_in[22];
  const float* ln1b  = (const float*)d_in[23];
  const float* w1w   = (const float*)d_in[24];
  const float* w1b   = (const float*)d_in[25];
  const float* ln2g  = (const float*)d_in[26];
  const float* ln2b  = (const float*)d_in[27];
  const float* w2w   = (const float*)d_in[28];
  const float* w2b   = (const float*)d_in[29];

  float* out    = (float*)d_out;          // (4,2048,64)
  float* newxyz = out + 4 * 2048 * 64;    // (4,2048,3)

  float* ws      = (float*)d_ws;
  float* xq      = ws;                    //   262144 f
  float* xkf     = xq  + 262144;          //  1048576 f
  float* xvf     = xkf + 1048576;         //  2097152 f
  float* pr1     = xvf + 2097152;         //   262144 f
  float* logits  = pr1 + 262144;          //  1048576 f
  float* smax    = logits + 1048576;      //      512 f
  float* ssum    = smax + 512;            //      512 f
  int*   ballidx = (int*)(ssum + 512);    //   131072 i   (~19 MB total)

  gemm_xq_kernel<<<128, 256, 0, stream>>>(x, wq_w, wq_b, xq);          // 1024 waves
  gemm_x1_kernel<<<512, 256, 0, stream>>>(x1, wk_w, wk_b, xkf, 1);     // 4096 waves
  gemm_x1_kernel<<<1024, 256, 0, stream>>>(x1, wv_w, wv_b, xvf, 2);    // 8192 waves
  fps_kernel<<<4, 512, 0, stream>>>(xyz1, newxyz);
  ball_kernel<<<1024, 256, 0, stream>>>(xyz1, newxyz, ballidx);
  pr1_kernel<<<32, 256, 0, stream>>>(xyz, p1l1w, p1l1b, p1lng, p1lnb, p1l2w, p1l2b, pr1);
  logits_kernel<<<512, 256, 0, stream>>>(xq, pr1, xkf, xyz1, newxyz, ballidx,
                                         pl1w, pl1b, plng, plnb, pl2w, pl2b,
                                         ln1g, ln1b, w1w, w1b, ln2g, ln2b, w2w, w2b,
                                         logits);
  softmax_stats_kernel<<<512, 256, 0, stream>>>(logits, smax, ssum);
  output_kernel<<<8192, 64, 0, stream>>>(xvf, ballidx, xyz1, newxyz,
                                         pl1w, pl1b, plng, plnb, pl2w, pl2b,
                                         logits, smax, ssum, out);
}
